// SequencePredictorGRU_39616778338682
// MI455X (gfx1250) — matrix operand order repaired and checked by reading, not compiled, not run on hardware
//
#include <hip/hip_runtime.h>
#include <stdint.h>

#define DEV __device__ __forceinline__

typedef __attribute__((ext_vector_type(16))) __bf16 v16bf;
typedef __attribute__((ext_vector_type(8)))  float  v8f;

union Frag {
    v16bf v;
    uint32_t u[8];
    uint4 q[2];
};

// round-to-nearest-even f32 -> bf16
DEV uint16_t bf16_rne(float f) {
    uint32_t u = __float_as_uint(f);
    return (uint16_t)((u + 0x7FFFu + ((u >> 16) & 1u)) >> 16);
}
DEV float bf16_tof(uint16_t h) { return __uint_as_float(((uint32_t)h) << 16); }

DEV uint32_t pack_bf2(float a, float b) {
    return (uint32_t)bf16_rne(a) | ((uint32_t)bf16_rne(b) << 16);
}
DEV float bf_lo(uint32_t u) { return __uint_as_float(u << 16); }
DEV float bf_hi(uint32_t u) { return __uint_as_float(u & 0xFFFF0000u); }

DEV v8f wmma_bf16(v16bf a, v16bf b, v8f c) {
    return __builtin_amdgcn_wmma_f32_16x16x32_bf16(
        false, a, false, b, (short)0, c, false, false);
}

DEV float wred(float v) {
#pragma unroll
    for (int m = 16; m > 0; m >>= 1) v += __shfl_xor(v, m, 32);
    return v;
}

DEV float gelu_exact(float x) {
    return 0.5f * x * (1.0f + erff(x * 0.70710678118654752f));
}

// ---------------------------------------------------------------------------
// Pack K x Ncols (B[k][n] = W[k*rowStride + n*colStride]) fp32 matrix into
// bf16 WMMA B-fragment order: tile (kt,nt) 32x16; dword = tile*256 + lane*8 + i
//   lane<16 : K = kt*32 + 2i,2i+1       N = nt*16 + lane
//   lane>=16: K = kt*32 + 16 + 2i,2i+1  N = nt*16 + lane-16
// ---------------------------------------------------------------------------
__global__ void pack_b_kernel(const float* __restrict__ W, uint32_t* __restrict__ out,
                              int K, int Ncols, int rowStride, int colStride) {
    const int nKT = K >> 5, nNT = Ncols >> 4;
    const int total = nKT * nNT * 256;
    for (int idx = blockIdx.x * blockDim.x + threadIdx.x; idx < total;
         idx += gridDim.x * blockDim.x) {
        int lanei = idx & 255;
        int tile  = idx >> 8;
        int nt = tile % nNT, kt = tile / nNT;
        int lane = lanei >> 3, i = lanei & 7;
        int n = nt * 16 + (lane & 15);
        int k = kt * 32 + ((lane >> 4) * 16) + 2 * i;
        float f0 = W[(size_t)k * rowStride + (size_t)n * colStride];
        float f1 = W[(size_t)(k + 1) * rowStride + (size_t)n * colStride];
        out[idx] = pack_bf2(f0, f1);
    }
}

// LDS row strides (dwords; %4==0 for 16B-aligned b128, pad breaks bank cycling)
#define LDXP  88    // packed X row: 80 dwords used (160 bf16)
#define LDA1P 520   // packed c1/act1 row: 512 dwords used (1024 bf16)
#define LDHP  264   // packed h row: 256 dwords used (512 bf16)
#define LDOF  264   // f32 o1 row: 256 used

// A-fragment from packed-bf16 LDS (row-major pairs). Tile at dword col kt*16.
DEV void loadAp(Frag& a, const uint32_t* __restrict__ tileBase, int ldaDw, int lane) {
    const int row = lane & 15;
    const int kh2 = (lane >> 4) * 8;
    const uint4* p = (const uint4*)(tileBase + row * ldaDw + kh2);
    a.q[0] = p[0];
    a.q[1] = p[1];   // +4 dwords = +8 bf16 columns
}

// Generic 32xN = (32xK packed-A) * packed-B gemm.
// MODE 0: f32 store of acc+bias               (Cout = float*,    ldc = f32 stride)
// MODE 1: bf16 b16-store of acc+bias          (Cout = uint16_t*, ldc in dwords)
// MODE 2: bf16 b16-store of tanh(acc+bias)    (Cout = uint16_t*, ldc in dwords)
template <int KT, int NT, int MODE>
DEV void gemm32p(const uint32_t* __restrict__ Ap, int ldaDw,
                 const uint32_t* __restrict__ Bp, const float* __restrict__ bias,
                 void* __restrict__ Cout, int ldc, int tid) {
    const int wave = tid >> 5, lane = tid & 31;
    const int row16 = lane & 15;
    for (int u = wave; u < 2 * NT; u += 8) {
        const int mt = u & 1, nt = u >> 1;
        v8f acc = {0.f, 0.f, 0.f, 0.f, 0.f, 0.f, 0.f, 0.f};
        for (int kt = 0; kt < KT; ++kt) {
            Frag a, b;
            loadAp(a, Ap + (mt * 16) * ldaDw + kt * 16, ldaDw, lane);
            const uint4* q = (const uint4*)(Bp + (size_t)(kt * NT + nt) * 256 + lane * 8);
            b.q[0] = q[0];
            b.q[1] = q[1];
            acc = wmma_bf16(a.v, b.v, acc);
        }
        const int col = nt * 16 + row16;
        const float bb = bias[col];
        const int r0 = mt * 16 + (lane >> 4) * 8;
        if (MODE == 0) {
            float* C = (float*)Cout;
#pragma unroll
            for (int v = 0; v < 8; ++v) C[(r0 + v) * ldc + col] = acc[v] + bb;
        } else {
            uint16_t* C = (uint16_t*)Cout;
            const int ldch = 2 * ldc;          // bf16 elements per row
#pragma unroll
            for (int v = 0; v < 8; ++v) {
                float val = acc[v] + bb;
                if (MODE == 2) val = tanhf(val);
                C[(r0 + v) * ldch + col] = bf16_rne(val);   // ds_store_b16, all lanes
            }
        }
    }
}

// ---------------------------------------------------------------------------
// Fused persistent kernel: 32 rows per block, init-MLP + T GRU steps.
// LDS (dwords): hAp[8448] hBp[8448] sX[32] | scratch R[19456]
//   init: Xp = R[0..2816) ; c1p/act1p = R[2816..19456) packed bf16 (LN in place)
//   step: o1f = R[0..8448) f32
// total 36384 dwords = 145,536 B  -> two workgroups fit per 320 KB WGP
// ---------------------------------------------------------------------------
__global__ __launch_bounds__(256) void gru_fused_kernel(
    const float* __restrict__ z_q, const float* __restrict__ f_prior,
    const float* __restrict__ b1, const float* __restrict__ g1,
    const float* __restrict__ be1, const float* __restrict__ b2,
    const float* __restrict__ start_token,
    const float* __restrict__ W_ih, const float* __restrict__ b_ih,
    const float* __restrict__ b_hh,
    const float* __restrict__ bo1, const float* __restrict__ go,
    const float* __restrict__ beo, const float* __restrict__ Wo2,
    const float* __restrict__ bo2,
    const uint32_t* __restrict__ W1p, const uint32_t* __restrict__ W2p,
    const uint32_t* __restrict__ Whhp, const uint32_t* __restrict__ Wo1p,
    float* __restrict__ preds, int T) {
    extern __shared__ uint32_t smem[];
    uint32_t* hAp = smem;                     // 8448
    uint32_t* hBp = smem + 8448;              // 8448
    float*    sX  = (float*)(smem + 16896);   // 32 (+pad)
    uint32_t* R   = smem + 16928;             // scratch (19456)

    uint32_t* Xp  = R;                        // init: packed X
    uint32_t* c1p = R + 2816;                 // init: packed c1 / act1 (in-place LN)
    float*    o1f = (float*)R;                // steps: f32 o1

    const int tid  = threadIdx.x;
    const int wave = tid >> 5, lane = tid & 31;
    const int row16 = lane & 15;
    const int n0 = blockIdx.x * 32;

    // ---- X = [z_q | f_prior], packed bf16 pairs (32 x 80 dwords) ----
    for (int idx = tid; idx < 32 * 80; idx += 256) {
        int r = idx / 80, d = idx - r * 80;
        int c = 2 * d;
        int gn = n0 + r;
        float v0 = (c < 128) ? z_q[(size_t)gn * 128 + c] : f_prior[(size_t)gn * 32 + (c - 128)];
        float v1 = (c + 1 < 128) ? z_q[(size_t)gn * 128 + c + 1]
                                 : f_prior[(size_t)gn * 32 + (c + 1 - 128)];
        Xp[r * LDXP + d] = pack_bf2(v0, v1);
    }
    __syncthreads();

    // ---- c1 = X @ W1 + b1  -> packed bf16 (32 x 512 dwords) ----
    gemm32p<5, 64, 1>(Xp, LDXP, W1p, b1, c1p, LDA1P, tid);
    __syncthreads();

    // ---- LayerNorm(1024) + exact GELU, packed in place (wave owns rows) ----
    for (int row = wave; row < 32; row += 8) {
        uint32_t* p = c1p + row * LDA1P;
        float s = 0.f;
        for (int j2 = lane; j2 < 512; j2 += 32) {
            uint32_t u = p[j2];
            s += bf_lo(u) + bf_hi(u);
        }
        s = wred(s);
        float mean = s * (1.0f / 1024.0f);
        float s2 = 0.f;
        for (int j2 = lane; j2 < 512; j2 += 32) {
            uint32_t u = p[j2];
            float d0 = bf_lo(u) - mean, d1 = bf_hi(u) - mean;
            s2 += d0 * d0 + d1 * d1;
        }
        s2 = wred(s2);
        float rstd = rsqrtf(s2 * (1.0f / 1024.0f) + 1e-5f);
        for (int j2 = lane; j2 < 512; j2 += 32) {
            uint32_t u = p[j2];
            int j = 2 * j2;
            float a0 = gelu_exact((bf_lo(u) - mean) * rstd * g1[j] + be1[j]);
            float a1 = gelu_exact((bf_hi(u) - mean) * rstd * g1[j + 1] + be1[j + 1]);
            p[j2] = pack_bf2(a0, a1);
        }
    }
    __syncthreads();

    // ---- h = tanh(act1 @ W2 + b2) -> packed hAp directly ----
    gemm32p<32, 32, 2>(c1p, LDA1P, W2p, b2, hAp, LDHP, tid);
    if (tid < 32) sX[tid] = start_token[0];
    __syncthreads();

    uint32_t* hA = hAp;
    uint32_t* hB = hBp;

    for (int t = 0; t < T; ++t) {
        uint16_t* hA16 = (uint16_t*)hA;
        uint16_t* hB16 = (uint16_t*)hB;
        // ---- GRU gates: r/z/n column-tile triple per wave unit ----
        for (int u = wave; u < 64; u += 8) {
            const int mt = u & 1, jt = u >> 1;
            v8f aR = {0.f, 0.f, 0.f, 0.f, 0.f, 0.f, 0.f, 0.f};
            v8f aZ = {0.f, 0.f, 0.f, 0.f, 0.f, 0.f, 0.f, 0.f};
            v8f aN = {0.f, 0.f, 0.f, 0.f, 0.f, 0.f, 0.f, 0.f};
            for (int kt = 0; kt < 16; ++kt) {
                Frag a, bR, bZ, bN;
                loadAp(a, hA + (mt * 16) * LDHP + kt * 16, LDHP, lane);
                const uint32_t* q = Whhp + (size_t)kt * 96 * 256 + lane * 8;
                const uint4* qR = (const uint4*)(q + (size_t)jt * 256);
                const uint4* qZ = (const uint4*)(q + (size_t)(32 + jt) * 256);
                const uint4* qN = (const uint4*)(q + (size_t)(64 + jt) * 256);
                bR.q[0] = qR[0]; bR.q[1] = qR[1];
                bZ.q[0] = qZ[0]; bZ.q[1] = qZ[1];
                bN.q[0] = qN[0]; bN.q[1] = qN[1];
                aR = wmma_bf16(a.v, bR.v, aR);
                aZ = wmma_bf16(a.v, bZ.v, aZ);
                aN = wmma_bf16(a.v, bN.v, aN);
            }
            const int col = jt * 16 + row16;
            const float wr = W_ih[col], wz = W_ih[512 + col], wn = W_ih[1024 + col];
            const float bir = b_ih[col], biz = b_ih[512 + col], bin_ = b_ih[1024 + col];
            const float bhr = b_hh[col], bhz = b_hh[512 + col], bhn = b_hh[1024 + col];
            const int r0 = mt * 16 + (lane >> 4) * 8;
#pragma unroll
            for (int v = 0; v < 8; ++v) {
                int row = r0 + v;
                float x = sX[row];
                float r = 1.f / (1.f + expf(-(x * wr + bir + aR[v] + bhr)));
                float z = 1.f / (1.f + expf(-(x * wz + biz + aZ[v] + bhz)));
                float n = tanhf(x * wn + bin_ + r * (aN[v] + bhn));
                float hp = bf16_tof(hA16[row * (2 * LDHP) + col]);   // ds_load_u16
                float hn = (1.f - z) * n + z * hp;
                hB16[row * (2 * LDHP) + col] = bf16_rne(hn);         // ds_store_b16
            }
        }
        __syncthreads();

        // ---- out MLP: o1 = h_new @ Wo1 + bo1 (32 x 256 f32) ----
        gemm32p<16, 16, 0>(hB, LDHP, Wo1p, bo1, o1f, LDOF, tid);
        __syncthreads();

        // ---- LN(256) + GELU + dot(Wo2) + bo2 -> preds, next x ----
        for (int row = wave; row < 32; row += 8) {
            const float* p = o1f + row * LDOF;
            float vals[8];
            float s = 0.f;
#pragma unroll
            for (int j = 0; j < 8; ++j) { vals[j] = p[lane + 32 * j]; s += vals[j]; }
            s = wred(s);
            float mean = s * (1.0f / 256.0f);
            float s2 = 0.f;
#pragma unroll
            for (int j = 0; j < 8; ++j) { float d = vals[j] - mean; s2 += d * d; }
            s2 = wred(s2);
            float rstd = rsqrtf(s2 * (1.0f / 256.0f) + 1e-5f);
            float y = 0.f;
#pragma unroll
            for (int j = 0; j < 8; ++j) {
                int c = lane + 32 * j;
                float v = (vals[j] - mean) * rstd * go[c] + beo[c];
                y += gelu_exact(v) * Wo2[c];
            }
            y = wred(y);
            if (lane == 0) {
                float out = y + bo2[0];
                sX[row] = out;
                preds[(size_t)(n0 + row) * T + t] = out;
            }
        }
        __syncthreads();

        uint32_t* tmp = hA; hA = hB; hB = tmp;
    }
}

// ---------------------------------------------------------------------------
// Workspace layout (dwords):
//   W1p  @ 0       : 5*64*256  = 81920    (K=160,  N=1024)
//   W2p  @ 81920   : 32*32*256 = 262144   (K=1024, N=512)
//   Whhp @ 344064  : 16*96*256 = 393216   (K=512,  N=1536, transposed W_hh)
//   Wo1p @ 737280  : 16*16*256 = 65536    (K=512,  N=256)
// ---------------------------------------------------------------------------
extern "C" void kernel_launch(void* const* d_in, const int* in_sizes, int n_in,
                              void* d_out, int out_size, void* d_ws, size_t ws_size,
                              hipStream_t stream) {
    const float* z_q   = (const float*)d_in[0];
    const float* f_pr  = (const float*)d_in[1];
    const float* W1    = (const float*)d_in[2];
    const float* b1    = (const float*)d_in[3];
    const float* g1    = (const float*)d_in[4];
    const float* be1   = (const float*)d_in[5];
    const float* W2    = (const float*)d_in[6];
    const float* b2    = (const float*)d_in[7];
    const float* stok  = (const float*)d_in[8];
    const float* W_ih  = (const float*)d_in[9];
    const float* W_hh  = (const float*)d_in[10];
    const float* b_ih  = (const float*)d_in[11];
    const float* b_hh  = (const float*)d_in[12];
    const float* Wo1   = (const float*)d_in[13];
    const float* bo1   = (const float*)d_in[14];
    const float* go    = (const float*)d_in[15];
    const float* beo   = (const float*)d_in[16];
    const float* Wo2   = (const float*)d_in[17];
    const float* bo2   = (const float*)d_in[18];
    float* preds = (float*)d_out;

    uint32_t* ws   = (uint32_t*)d_ws;
    uint32_t* W1p  = ws;
    uint32_t* W2p  = ws + 81920;
    uint32_t* Whhp = ws + 344064;
    uint32_t* Wo1p = ws + 737280;

    const int N = in_sizes[0] / 128;
    const int T = 20;

    pack_b_kernel<<<256, 256, 0, stream>>>(W1,   W1p,  160,  1024, 1024, 1);
    pack_b_kernel<<<512, 256, 0, stream>>>(W2,   W2p,  1024, 512,  512,  1);
    pack_b_kernel<<<512, 256, 0, stream>>>(W_hh, Whhp, 512,  1536, 1,    512);
    pack_b_kernel<<<256, 256, 0, stream>>>(Wo1,  Wo1p, 512,  256,  256,  1);

    const size_t shbytes = (size_t)36384 * sizeof(uint32_t); // 145,536 B
    hipFuncSetAttribute((const void*)gru_fused_kernel,
                        hipFuncAttributeMaxDynamicSharedMemorySize, (int)shbytes);

    gru_fused_kernel<<<N / 32, 256, shbytes, stream>>>(
        z_q, f_pr, b1, g1, be1, b2, stok, W_ih, b_ih, b_hh,
        bo1, go, beo, Wo2, bo2, W1p, W2p, Whhp, Wo1p, preds, T);
}
